// HopfieldMemoryLayer_6090263626370
// MI455X (gfx1250) — compile-verified, matrix operand order checked by reading, not run on hardware
//
#include <hip/hip_runtime.h>

// ---- dims fixed by the reference -------------------------------------------
static constexpr int Bc = 2, Sc = 2048, Ec = 2048, Hc = 16, Mc = 2048, Dc = 128;

typedef __attribute__((ext_vector_type(16))) __bf16 v16bf;
typedef __attribute__((ext_vector_type(8)))  __bf16 v8bf;
typedef __attribute__((ext_vector_type(8)))  float  v8f;
typedef __attribute__((address_space(3)))    __bf16 as3_bf16;

#define DEV __device__ __forceinline__

DEV v8f v8f_zero() { v8f z = {0.f,0.f,0.f,0.f,0.f,0.f,0.f,0.f}; return z; }

// Load a 16x32 bf16 A-fragment from a row-major matrix (ld in elements).
// Lane layout (ISA 7.12.2, 16-bit A 16x32): row = lane&15, element i maps to
// K = (i>>3)*16 + (lane>>4)*8 + (i&7)  -> two contiguous 16B loads per lane.
// B operands are fed by loading B^T rows with this same pattern (symmetric
// A/B lane layouts for 16-bit WMMA).
DEV v16bf load_frag(const __bf16* base, int row0, int k0, int ld) {
  const int lane = threadIdx.x & 31;
  const int row  = row0 + (lane & 15);
  const int kh   = k0 + ((lane >> 4) << 3);
  const __bf16* p = base + (size_t)row * ld + kh;
  v8bf lo = *reinterpret_cast<const v8bf*>(p);
  v8bf hi = *reinterpret_cast<const v8bf*>(p + 16);
  return __builtin_shufflevector(lo, hi, 0,1,2,3,4,5,6,7,8,9,10,11,12,13,14,15);
}

DEV v8f wmma_bf16(v16bf a, v16bf b, v8f c) {
  // v_wmma_f32_16x16x32_bf16
  return __builtin_amdgcn_wmma_f32_16x16x32_bf16(false, a, false, b,
                                                 (short)0, c, false, false);
}

// Async global -> LDS copy of 16 bytes per lane (GLOBAL_LOAD_ASYNC_TO_LDS_B128,
// GV addressing, tracked by ASYNCcnt).
DEV void async_copy_b128(__bf16* ldst, const __bf16* src) {
  const unsigned loff = (unsigned)(size_t)(as3_bf16*)ldst;  // LDS byte offset
  asm volatile("global_load_async_to_lds_b128 %0, %1, off"
               :: "v"(loff), "v"(src)
               : "memory");
}

// reductions across the 16 lanes sharing a C/D-tile row (halves of the wave)
DEV float hmax16(float x) {
  x = fmaxf(x, __shfl_xor(x, 1));
  x = fmaxf(x, __shfl_xor(x, 2));
  x = fmaxf(x, __shfl_xor(x, 4));
  x = fmaxf(x, __shfl_xor(x, 8));
  return x;
}
DEV float hsum16(float x) {
  x += __shfl_xor(x, 1);
  x += __shfl_xor(x, 2);
  x += __shfl_xor(x, 4);
  x += __shfl_xor(x, 8);
  return x;
}

// ---- kernel 1: RMSNorm(query) -> bf16 --------------------------------------
__global__ __launch_bounds__(256) void k_rms_q(const float* __restrict__ x,
                                               const float* __restrict__ w,
                                               __bf16* __restrict__ out) {
  const int row = blockIdx.x, tid = threadIdx.x;
  const float* xr = x + (size_t)row * Ec;
  float ss = 0.f;
  #pragma unroll
  for (int i = 0; i < Ec / 256; ++i) { float v = xr[tid + i*256]; ss += v*v; }
  #pragma unroll
  for (int off = 16; off; off >>= 1) ss += __shfl_xor(ss, off);
  __shared__ float red[8];
  if ((tid & 31) == 0) red[tid >> 5] = ss;
  __syncthreads();
  if (tid < 8) {
    float v = red[tid];
    #pragma unroll
    for (int off = 4; off; off >>= 1) v += __shfl_xor(v, off);
    if (tid == 0) red[0] = v;
  }
  __syncthreads();
  const float inv = rsqrtf(red[0] * (1.0f / Ec) + 1e-5f);
  #pragma unroll
  for (int i = 0; i < Ec / 256; ++i) {
    int j = tid + i*256;
    out[(size_t)row * Ec + j] = (__bf16)(xr[j] * inv * w[j]);
  }
}

// ---- kernel 2: f32 -> bf16 elementwise (wq) --------------------------------
__global__ __launch_bounds__(256) void k_cvt_bf16(const float* __restrict__ in,
                                                  __bf16* __restrict__ out, int n) {
  int i = blockIdx.x * 256 + threadIdx.x;
  if (i < n) out[i] = (__bf16)in[i];
}

// ---- kernel 3: per-head K = P*wk^T (row-major), V^T = (P*wv^T)^T -----------
__global__ __launch_bounds__(256) void k_kv(const float* __restrict__ pat,
                                            const float* __restrict__ wk,
                                            const float* __restrict__ wv,
                                            __bf16* __restrict__ kout,
                                            __bf16* __restrict__ vTout) {
  int t = blockIdx.x * 256 + threadIdx.x;
  if (t >= Hc * Mc * Dc) return;
  int e  = t & (Dc - 1);
  int hm = t >> 7;                 // Dc = 128
  int m  = hm & (Mc - 1);
  int h  = hm >> 11;               // Mc = 2048
  const float* pr = pat + ((size_t)h * Mc + m) * Dc;
  const float* kr = wk + (size_t)e * Dc;
  const float* vr = wv + (size_t)e * Dc;
  float ak = 0.f, av = 0.f;
  #pragma unroll 8
  for (int i = 0; i < Dc; ++i) { float p = pr[i]; ak += p * kr[i]; av += p * vr[i]; }
  kout[((size_t)h * Mc + m) * Dc + e]  = (__bf16)ak;
  vTout[((size_t)h * Dc + e) * Mc + m] = (__bf16)av;
}

// ---- kernel 4: q = qn @ wq^T, stored as bf16 [B,H,S,D] ---------------------
__global__ __launch_bounds__(128) void k_qproj(const __bf16* __restrict__ qn,
                                               const __bf16* __restrict__ wqb,
                                               __bf16* __restrict__ qout) {
  const int w = threadIdx.x >> 5, lane = threadIdx.x & 31;
  const int halfw = lane >> 4, col = lane & 15;
  const int row0 = blockIdx.x * 16;          // token tile
  const int col0 = blockIdx.y * 64 + w * 16; // output-feature tile
  v8f acc = v8f_zero();
  for (int k0 = 0; k0 < Ec; k0 += 32) {
    v16bf a = load_frag(qn,  row0, k0, Ec);
    v16bf b = load_frag(wqb, col0, k0, Ec);  // wq rows == (wq^T) columns
    acc = wmma_bf16(a, b, acc);
  }
  const int f = col0 + col;                  // stays inside one head (D=128 % 16 == 0)
  const int h = f >> 7, d = f & 127;
  #pragma unroll
  for (int r = 0; r < 8; ++r) {
    int tok = row0 + r + 8 * halfw;          // C/D layout: row = r + 8*half
    int b_ = tok >> 11, s = tok & (Sc - 1);
    qout[(((size_t)b_ * Hc + h) * Sc + s) * Dc + d] = (__bf16)acc[r];
  }
}

// ---- kernel 5: flash attention over memory slots ---------------------------
// 4 waves / block, each wave owns 16 query rows. K/V^T chunks (32 slots) are
// staged once per block into padded LDS tiles with GLOBAL_LOAD_ASYNC_TO_LDS
// (double-buffered: DMA of chunk i+1 overlaps WMMA of chunk i; ASYNCcnt
// returns in order, so waiting <=8 after issuing the next stage's 8 ops
// guarantees the current stage is resident). Padded LDS strides (272B / 80B
// per row) make the 16-lane ds_load_b128 fragment reads bank-conflict-free.
static constexpr int KLD = 136;   // 128 + 8 pad (bf16 elements)
static constexpr int VLD = 40;    // 32 + 8 pad
static constexpr int PLD = 40;    // 32 + 8 pad

DEV void stage_chunk(const __bf16* kbase, const __bf16* vbase, int mo,
                     __bf16* kdst, __bf16* vdst, int tid) {
  // K chunk: 32 rows x 128 bf16 (contiguous rows) -> kdst rows of KLD
  #pragma unroll
  for (int i = 0; i < 4; ++i) {
    int idx = tid + i * 128;                 // 0..511 b128 transfers
    int row = idx >> 4, part = idx & 15;
    async_copy_b128(kdst + row * KLD + part * 8,
                    kbase + (size_t)(mo + row) * Dc + part * 8);
  }
  // V^T chunk: 128 rows x 32 bf16 (row stride Mc) -> vdst rows of VLD
  #pragma unroll
  for (int i = 0; i < 4; ++i) {
    int idx = tid + i * 128;                 // 0..511 b128 transfers
    int d = idx >> 2, part = idx & 3;
    async_copy_b128(vdst + d * VLD + part * 8,
                    vbase + (size_t)d * Mc + mo + part * 8);
  }
}

__global__ __launch_bounds__(128) void k_attn(const __bf16* __restrict__ qb,
                                              const __bf16* __restrict__ kb,
                                              const __bf16* __restrict__ vT,
                                              const float* __restrict__ beta,
                                              float* __restrict__ retr) {
  __shared__ alignas(16) __bf16 kbuf[2][32 * KLD];
  __shared__ alignas(16) __bf16 vbuf[2][128 * VLD];
  __shared__ alignas(16) __bf16 pbuf[4][16 * PLD];
  const int tid = threadIdx.x;
  const int w = tid >> 5, lane = tid & 31;
  const int halfw = lane >> 4, col = lane & 15;
  const int h = blockIdx.y, b = blockIdx.z;
  const int row0 = blockIdx.x * 64 + w * 16;
  const __bf16* qbase = qb + ((size_t)b * Hc + h) * Sc * Dc;
  const __bf16* kbase = kb + (size_t)h * Mc * Dc;
  const __bf16* vbase = vT + (size_t)h * Dc * Mc;
  __bf16* pb = &pbuf[w][0];
  const float sc = beta[h] * 0.08838834764831845f; // 1/sqrt(128)

  // stage first chunk while we load q fragments from global
  stage_chunk(kbase, vbase, 0, kbuf[0], vbuf[0], tid);

  v16bf qa[4];
  #pragma unroll
  for (int j = 0; j < 4; ++j) qa[j] = load_frag(qbase, row0, j * 32, Dc);

  float mrow[8], lrow[8];
  v8f O[8];
  #pragma unroll
  for (int r = 0; r < 8; ++r) { mrow[r] = -3.0e38f; lrow[r] = 0.f; O[r] = v8f_zero(); }

  for (int mo = 0; mo < Mc; mo += 32) {
    const int cur = (mo >> 5) & 1;
    if (mo + 32 < Mc) {
      stage_chunk(kbase, vbase, mo + 32, kbuf[cur ^ 1], vbuf[cur ^ 1], tid);
      asm volatile("s_wait_asynccnt 0x8" ::: "memory");  // current stage resident
    } else {
      asm volatile("s_wait_asynccnt 0x0" ::: "memory");
    }
    __syncthreads();
    const __bf16* kc = kbuf[cur];
    const __bf16* vc = vbuf[cur];

    // scores: two 16-col tiles, K-loop over D=128, operands from LDS
    v8f s0 = v8f_zero(), s1 = v8f_zero();
    #pragma unroll
    for (int j = 0; j < 4; ++j) s0 = wmma_bf16(qa[j], load_frag(kc, 0,  j * 32, KLD), s0);
    #pragma unroll
    for (int j = 0; j < 4; ++j) s1 = wmma_bf16(qa[j], load_frag(kc, 16, j * 32, KLD), s1);

    float c[8];
    #pragma unroll
    for (int r = 0; r < 8; ++r) {
      float x0 = s0[r] * sc, x1 = s1[r] * sc;
      float nm = fmaxf(mrow[r], hmax16(fmaxf(x0, x1)));
      float cc = __expf(mrow[r] - nm);
      float p0 = __expf(x0 - nm), p1 = __expf(x1 - nm);
      lrow[r] = lrow[r] * cc + hsum16(p0 + p1);
      mrow[r] = nm; c[r] = cc;
      pb[(r + 8 * halfw) * PLD + col]      = (__bf16)p0;  // C-layout -> row-major LDS
      pb[(r + 8 * halfw) * PLD + 16 + col] = (__bf16)p1;
    }
    #pragma unroll
    for (int t = 0; t < 8; ++t) {
      #pragma unroll
      for (int r = 0; r < 8; ++r) O[t][r] *= c[r];
    }
    asm volatile("s_wait_dscnt 0x0" ::: "memory");        // LDS RAW fence (same wave)
    v16bf pa = load_frag(pb, 0, 0, PLD);                  // reload as A-fragment
    #pragma unroll
    for (int t = 0; t < 8; ++t)
      O[t] = wmma_bf16(pa, load_frag(vc, t * 16, 0, VLD), O[t]);
    __syncthreads();   // all waves done reading buf[cur] before it is re-staged
  }

  #pragma unroll
  for (int t = 0; t < 8; ++t) {
    const int d = t * 16 + col;
    #pragma unroll
    for (int r = 0; r < 8; ++r) {
      const int srow = row0 + r + 8 * halfw;
      retr[((size_t)b * Sc + srow) * (Hc * Dc) + h * Dc + d] = O[t][r] / lrow[r];
    }
  }
}

// ---- kernel 6: out = query + RMSNorm(retrieved) * norm_r_w -----------------
__global__ __launch_bounds__(256) void k_rms_add(const float* __restrict__ r,
                                                 const float* __restrict__ w,
                                                 const float* __restrict__ q,
                                                 float* __restrict__ out) {
  const int row = blockIdx.x, tid = threadIdx.x;
  const float* xr = r + (size_t)row * Ec;
  float ss = 0.f;
  #pragma unroll
  for (int i = 0; i < Ec / 256; ++i) { float v = xr[tid + i*256]; ss += v*v; }
  #pragma unroll
  for (int off = 16; off; off >>= 1) ss += __shfl_xor(ss, off);
  __shared__ float red[8];
  if ((tid & 31) == 0) red[tid >> 5] = ss;
  __syncthreads();
  if (tid < 8) {
    float v = red[tid];
    #pragma unroll
    for (int off = 4; off; off >>= 1) v += __shfl_xor(v, off);
    if (tid == 0) red[0] = v;
  }
  __syncthreads();
  const float inv = rsqrtf(red[0] * (1.0f / Ec) + 1e-5f);
  #pragma unroll
  for (int i = 0; i < Ec / 256; ++i) {
    int j = tid + i*256;
    out[(size_t)row * Ec + j] = q[(size_t)row * Ec + j] + xr[j] * inv * w[j];
  }
}

// ---- host launcher ---------------------------------------------------------
extern "C" void kernel_launch(void* const* d_in, const int* in_sizes, int n_in,
                              void* d_out, int out_size, void* d_ws, size_t ws_size,
                              hipStream_t stream) {
  (void)in_sizes; (void)n_in; (void)out_size; (void)ws_size;
  const float* query    = (const float*)d_in[0];
  const float* patterns = (const float*)d_in[1];
  const float* wq       = (const float*)d_in[2];
  const float* wk       = (const float*)d_in[3];
  const float* wv       = (const float*)d_in[4];
  const float* beta     = (const float*)d_in[5];
  const float* nqw      = (const float*)d_in[6];
  const float* nrw      = (const float*)d_in[7];
  float* out = (float*)d_out;

  char* ws = (char*)d_ws;
  size_t off = 0;
  __bf16* qn_bf = (__bf16*)(ws + off); off += (size_t)Bc * Sc * Ec * 2;       // 16.8 MB
  __bf16* wq_bf = (__bf16*)(ws + off); off += (size_t)Hc * Dc * Ec * 2;       //  8.4 MB
  __bf16* q_bf  = (__bf16*)(ws + off); off += (size_t)Bc * Hc * Sc * Dc * 2;  // 16.8 MB
  __bf16* k_bf  = (__bf16*)(ws + off); off += (size_t)Hc * Mc * Dc * 2;       //  8.4 MB
  __bf16* vT_bf = (__bf16*)(ws + off); off += (size_t)Hc * Dc * Mc * 2;       //  8.4 MB
  float*  retr  = (float*)(ws + off);  off += (size_t)Bc * Sc * Hc * Dc * 4;  // 33.6 MB

  k_rms_q  <<<Bc * Sc, 256, 0, stream>>>(query, nqw, qn_bf);
  k_cvt_bf16<<<(Hc * Dc * Ec + 255) / 256, 256, 0, stream>>>(wq, wq_bf, Hc * Dc * Ec);
  k_kv     <<<(Hc * Mc * Dc + 255) / 256, 256, 0, stream>>>(patterns, wk, wv, k_bf, vT_bf);
  k_qproj  <<<dim3((Bc * Sc) / 16, (Hc * Dc) / 64), 128, 0, stream>>>(qn_bf, wq_bf, q_bf);
  k_attn   <<<dim3(Sc / 64, Hc, Bc), 128, 0, stream>>>(q_bf, k_bf, vT_bf, beta, retr);
  k_rms_add<<<Bc * Sc, 256, 0, stream>>>(retr, nrw, query, out);
}
